// GraphSAGE_6373731468071
// MI455X (gfx1250) — matrix operand order repaired and checked by reading, not compiled, not run on hardware
//
#include <hip/hip_runtime.h>
#include <hip/hip_bf16.h>

typedef __attribute__((ext_vector_type(16))) __bf16 v16bf;
typedef __attribute__((ext_vector_type(8)))  float  v8f;

#define N_NODES 50000
#define N_EDGES 800000
#define F 128
#define ROWS_PER_BLOCK 256   // 16 strips of 16 rows; 8 waves, one n-tile each
#define STRIPS (ROWS_PER_BLOCK / 16)

union FragBF { v16bf v; uint4 q[2]; };

// ---- WMMA fragment loaders (bf16, 16x16x32, wave32 layouts per CDNA5 ISA 7.12.2) ----

// A: 16x32 tile from row-major bf16 [rows x 128].
// lane 0-15: m=lane,   K in {k0..k0+7, k0+16..k0+23}
// lane16-31: m=lane-16,K in {k0+8..k0+15, k0+24..k0+31}
__device__ __forceinline__ v16bf load_a_frag(const __hip_bfloat16* __restrict__ A, int m, int k0, int lane) {
    const int kb = k0 + ((lane >> 4) << 3);
    const __hip_bfloat16* p = A + (size_t)m * F + kb;
    FragBF f;
    f.q[0] = *reinterpret_cast<const uint4*>(p);        // 8 bf16: kb .. kb+7
    f.q[1] = *reinterpret_cast<const uint4*>(p + 16);   // 8 bf16: kb+16 .. kb+23
    return f.v;
}

// B: 32x16 tile. Wt is the TRANSPOSED weight, row-major [n][k] (128x128 bf16).
// lane 0-15: n=lane, K in {k0..k0+7, k0+16..k0+23} ; lane16-31: n=lane-16, K in {k0+8..k0+15, k0+24..k0+31}
__device__ __forceinline__ v16bf load_b_frag(const __hip_bfloat16* __restrict__ Wt, int k0, int n0, int lane) {
    const int n = n0 + (lane & 15);
    const int k = k0 + ((lane >> 4) << 3);
    const __hip_bfloat16* p = Wt + (size_t)n * F + k;
    FragBF f;
    f.q[0] = *reinterpret_cast<const uint4*>(p);        // K = k .. k+7
    f.q[1] = *reinterpret_cast<const uint4*>(p + 16);   // K = k+16 .. k+23
    return f.v;
}

// Load one 16-row strip's A fragments (both matrices) into registers.
__device__ __forceinline__ void load_strip(const __hip_bfloat16* __restrict__ Am,
                                           const __hip_bfloat16* __restrict__ Ax,
                                           int row0, int nrows, int lane,
                                           v16bf am[4], v16bf ax[4]) {
    int mload = row0 + (lane & 15);
    if (mload > nrows - 1) mload = nrows - 1;           // clamp: keep EXEC all-ones for WMMA
    int mpre = mload + 32;                               // prefetch two strips ahead
    if (mpre > nrows - 1) mpre = nrows - 1;
    __builtin_prefetch(Am + (size_t)mpre * F, 0, 1);
    __builtin_prefetch(Ax + (size_t)mpre * F, 0, 1);
#pragma unroll
    for (int kt = 0; kt < 4; ++kt) {
        am[kt] = load_a_frag(Am, mload, kt * 32, lane);
        ax[kt] = load_a_frag(Ax, mload, kt * 32, lane);
    }
}

// 8-WMMA chain + bias + relu + store for one strip.
__device__ __forceinline__ void compute_store(const v16bf am[4], const v16bf ax[4],
                                              const v16bf bl[4], const v16bf br[4],
                                              float bv, int row0, int nrows, int lane, int n0,
                                              float* __restrict__ Hf,
                                              __hip_bfloat16* __restrict__ Hbf) {
    v8f c = {};
#pragma unroll
    for (int kt = 0; kt < 4; ++kt) {
        c = __builtin_amdgcn_wmma_f32_16x16x32_bf16(false, am[kt], false, bl[kt],
                                                    (short)0, c, false, false);
        c = __builtin_amdgcn_wmma_f32_16x16x32_bf16(false, ax[kt], false, br[kt],
                                                    (short)0, c, false, false);
    }
    const int ncol = lane & 15;
    const int mbase = row0 + ((lane >> 4) << 3);        // C layout: m=row0+(lane>>4)*8+i
    const size_t obase = (size_t)mbase * F + n0 + ncol;
    float r[8];
#pragma unroll
    for (int i = 0; i < 8; ++i) {
        const float v = c[i] + bv;
        r[i] = v > 0.f ? v : 0.f;
    }
    if (row0 + 16 <= nrows) {                           // uniform fast path
#pragma unroll
        for (int i = 0; i < 8; ++i) Hf[obase + (size_t)i * F] = r[i];
        if (Hbf) {
#pragma unroll
            for (int i = 0; i < 8; ++i) Hbf[obase + (size_t)i * F] = __float2bfloat16(r[i]);
        }
    } else {                                            // boundary strip only
#pragma unroll
        for (int i = 0; i < 8; ++i)
            if (mbase + i < nrows) Hf[obase + (size_t)i * F] = r[i];
        if (Hbf) {
#pragma unroll
            for (int i = 0; i < 8; ++i)
                if (mbase + i < nrows) Hbf[obase + (size_t)i * F] = __float2bfloat16(r[i]);
        }
    }
}

// ---- Fused dual-GEMM: H = relu(Am @ Wl + bias + Ax @ Wr), bf16 in, f32 accum ----
// Weights register-resident; A strips double-buffered (ping-pong, unroll-by-2) so the
// next strip's global loads are in flight while the current strip's WMMA chain runs.
__global__ void __launch_bounds__(256)
sage_gemm_wmma(const __hip_bfloat16* __restrict__ Am,
               const __hip_bfloat16* __restrict__ Wl_t,
               const float* __restrict__ bias,
               const __hip_bfloat16* __restrict__ Ax,
               const __hip_bfloat16* __restrict__ Wr_t,
               float* __restrict__ Hf,
               __hip_bfloat16* __restrict__ Hbf,
               int nrows) {
    const int lane = threadIdx.x & 31;
    const int wave = threadIdx.x >> 5;
    const int n0   = wave * 16;                 // this wave's output column tile

    // B fragments: resident for the whole kernel (4 k-tiles x 2 matrices = 64 VGPRs)
    v16bf bl[4], br[4];
#pragma unroll
    for (int kt = 0; kt < 4; ++kt) {
        bl[kt] = load_b_frag(Wl_t, kt * 32, n0, lane);
        br[kt] = load_b_frag(Wr_t, kt * 32, n0, lane);
    }
    const float bv = bias[n0 + (lane & 15)];

    const int rbase = blockIdx.x * ROWS_PER_BLOCK;
    const int remaining = nrows - rbase;
    if (remaining <= 0) return;
    int nstrips = (remaining + 15) >> 4;
    if (nstrips > STRIPS) nstrips = STRIPS;

    v16bf amA[4], axA[4], amB[4], axB[4];
    load_strip(Am, Ax, rbase, nrows, lane, amA, axA);

#pragma unroll 1
    for (int s = 0; s < nstrips; s += 2) {
        if (s + 1 < nstrips)
            load_strip(Am, Ax, rbase + (s + 1) * 16, nrows, lane, amB, axB);
        compute_store(amA, axA, bl, br, bv, rbase + s * 16, nrows, lane, n0, Hf, Hbf);
        if (s + 2 < nstrips)
            load_strip(Am, Ax, rbase + (s + 2) * 16, nrows, lane, amA, axA);
        if (s + 1 < nstrips)
            compute_store(amB, axB, bl, br, bv, rbase + (s + 1) * 16, nrows, lane, n0, Hf, Hbf);
    }
}

// ---- Irregular graph kernels ----

__global__ void sage_degree(const int* __restrict__ dst, float* __restrict__ deg, int nedges) {
    const int e = blockIdx.x * blockDim.x + threadIdx.x;
    if (e < nedges) atomicAdd(&deg[dst[e]], 1.0f);
}

// 32 lanes per edge, 4 consecutive floats per lane (float4 read, 4 f32 atomic adds).
__global__ void sage_aggregate(const float* __restrict__ X,
                               const int* __restrict__ src,
                               const int* __restrict__ dst,
                               float* __restrict__ agg, int nedges) {
    const int gid = blockIdx.x * blockDim.x + threadIdx.x;
    const int e = gid >> 5;
    if (e >= nedges) return;
    const int fo = (gid & 31) << 2;
    const int s = src[e];
    const int d = dst[e];
    const float4 v = *reinterpret_cast<const float4*>(X + (size_t)s * F + fo);
    float* p = agg + (size_t)d * F + fo;
    atomicAdd(p + 0, v.x);
    atomicAdd(p + 1, v.y);
    atomicAdd(p + 2, v.z);
    atomicAdd(p + 3, v.w);
}

// mean = agg / max(deg,1), emitted directly as bf16 for the WMMA GEMM.
__global__ void sage_normalize(const float* __restrict__ agg, const float* __restrict__ deg,
                               __hip_bfloat16* __restrict__ meanbf, int nnodes) {
    const int gid = blockIdx.x * blockDim.x + threadIdx.x;
    const int nd = gid >> 5;
    if (nd >= nnodes) return;
    const int fo = (gid & 31) << 2;
    const float inv = 1.0f / fmaxf(deg[nd], 1.0f);
    const float4 v = *reinterpret_cast<const float4*>(agg + (size_t)nd * F + fo);
    __hip_bfloat16* p = meanbf + (size_t)nd * F + fo;
    p[0] = __float2bfloat16(v.x * inv);
    p[1] = __float2bfloat16(v.y * inv);
    p[2] = __float2bfloat16(v.z * inv);
    p[3] = __float2bfloat16(v.w * inv);
}

__global__ void f32_to_bf16(const float* __restrict__ in, __hip_bfloat16* __restrict__ out, int n) {
    const int i = blockIdx.x * blockDim.x + threadIdx.x;
    if (i < n) out[i] = __float2bfloat16(in[i]);
}

// Wt[n*128+k] = bf16(W[k*128+n])  (128x128)
__global__ void w_transpose_bf16(const float* __restrict__ W, __hip_bfloat16* __restrict__ Wt) {
    const int i = blockIdx.x * blockDim.x + threadIdx.x;
    const int k = i >> 7;
    const int n = i & 127;
    Wt[(size_t)n * F + k] = __float2bfloat16(W[(size_t)k * F + n]);
}

// logits = h @ Wlin + blin ; out = log_softmax (2 classes)
__global__ void sage_head(const float* __restrict__ H, const float* __restrict__ Wlin,
                          const float* __restrict__ blin, float* __restrict__ out, int nnodes) {
    const int i = blockIdx.x * blockDim.x + threadIdx.x;
    if (i >= nnodes) return;
    const float* h = H + (size_t)i * F;
    float a0 = blin[0], a1 = blin[1];
#pragma unroll 4
    for (int k = 0; k < F; ++k) {
        const float v = h[k];
        a0 += v * Wlin[2 * k];
        a1 += v * Wlin[2 * k + 1];
    }
    const float mx = fmaxf(a0, a1);
    const float lse = mx + __logf(__expf(a0 - mx) + __expf(a1 - mx));
    out[2 * (size_t)i + 0] = a0 - lse;
    out[2 * (size_t)i + 1] = a1 - lse;
}

extern "C" void kernel_launch(void* const* d_in, const int* in_sizes, int n_in,
                              void* d_out, int out_size, void* d_ws, size_t ws_size,
                              hipStream_t stream) {
    (void)in_sizes; (void)n_in; (void)out_size; (void)ws_size;
    const float* x    = (const float*)d_in[0];
    const int*   ei   = (const int*)d_in[1];     // [2, E]: row 0 = src, row 1 = dst
    const float* W1l  = (const float*)d_in[2];
    const float* b1   = (const float*)d_in[3];
    const float* W1r  = (const float*)d_in[4];
    const float* W2l  = (const float*)d_in[5];
    const float* b2   = (const float*)d_in[6];
    const float* W2r  = (const float*)d_in[7];
    const float* Wlin = (const float*)d_in[8];
    const float* blin = (const float*)d_in[9];
    float* out = (float*)d_out;

    const int* src = ei;
    const int* dst = ei + N_EDGES;

    // ---- workspace carve-up (256B aligned) ----
    char* ws = (char*)d_ws;
    size_t off = 0;
    auto carve = [&](size_t bytes) { char* p = ws + off; off = (off + bytes + 255) & ~(size_t)255; return p; };
    float*          deg    = (float*)carve((size_t)N_NODES * 4);
    float*          agg    = (float*)carve((size_t)N_NODES * F * 4);  // reused both layers
    float*          h1f    = (float*)carve((size_t)N_NODES * F * 4);  // layer1 f32 out; reused as h2
    __hip_bfloat16* xbf    = (__hip_bfloat16*)carve((size_t)N_NODES * F * 2);
    __hip_bfloat16* meanbf = (__hip_bfloat16*)carve((size_t)N_NODES * F * 2); // reused both layers
    __hip_bfloat16* h1bf   = (__hip_bfloat16*)carve((size_t)N_NODES * F * 2);
    __hip_bfloat16* W1l_t  = (__hip_bfloat16*)carve((size_t)F * F * 2);
    __hip_bfloat16* W1r_t  = (__hip_bfloat16*)carve((size_t)F * F * 2);
    __hip_bfloat16* W2l_t  = (__hip_bfloat16*)carve((size_t)F * F * 2);
    __hip_bfloat16* W2r_t  = (__hip_bfloat16*)carve((size_t)F * F * 2);

    const int nfeat = N_NODES * F;                 // 6,400,000
    const int aggThreads = N_EDGES * 32;           // 25,600,000
    const int normThreads = N_NODES * 32;          // 1,600,000
    const dim3 B(256);
    const int gemmBlocks = (N_NODES + ROWS_PER_BLOCK - 1) / ROWS_PER_BLOCK;  // 196

    // zero accumulators
    hipMemsetAsync(deg, 0, (size_t)N_NODES * 4, stream);
    hipMemsetAsync(agg, 0, (size_t)N_NODES * F * 4, stream);

    // precompute: bf16 x, transposed bf16 weights, degrees
    f32_to_bf16<<<(nfeat + 255) / 256, B, 0, stream>>>(x, xbf, nfeat);
    w_transpose_bf16<<<(F * F) / 256, B, 0, stream>>>(W1l, W1l_t);
    w_transpose_bf16<<<(F * F) / 256, B, 0, stream>>>(W1r, W1r_t);
    w_transpose_bf16<<<(F * F) / 256, B, 0, stream>>>(W2l, W2l_t);
    w_transpose_bf16<<<(F * F) / 256, B, 0, stream>>>(W2r, W2r_t);
    sage_degree<<<(N_EDGES + 255) / 256, B, 0, stream>>>(dst, deg, N_EDGES);

    // ---- layer 1 ----
    sage_aggregate<<<(aggThreads + 255) / 256, B, 0, stream>>>(x, src, dst, agg, N_EDGES);
    sage_normalize<<<(normThreads + 255) / 256, B, 0, stream>>>(agg, deg, meanbf, N_NODES);
    sage_gemm_wmma<<<gemmBlocks, B, 0, stream>>>(meanbf, W1l_t, b1, xbf, W1r_t, h1f, h1bf, N_NODES);

    // ---- layer 2 ----
    hipMemsetAsync(agg, 0, (size_t)N_NODES * F * 4, stream);
    sage_aggregate<<<(aggThreads + 255) / 256, B, 0, stream>>>(h1f, src, dst, agg, N_EDGES);
    sage_normalize<<<(normThreads + 255) / 256, B, 0, stream>>>(agg, deg, meanbf, N_NODES);
    sage_gemm_wmma<<<gemmBlocks, B, 0, stream>>>(meanbf, W2l_t, b2, h1bf, W2r_t, h1f, nullptr, N_NODES);

    // ---- head ----
    sage_head<<<(N_NODES + 255) / 256, B, 0, stream>>>(h1f, Wlin, blin, out, N_NODES);
}
